// DisentangledSpatialAttention_64561948394161
// MI455X (gfx1250) — compile-verified, matrix operand order checked by reading, not run on hardware
//
#include <hip/hip_runtime.h>

#define B_ 2
#define L_ 2048
#define E_ 1024
#define H_ 16
#define D_ 64
#define R_ (B_ * L_)   // 4096 token rows

typedef __attribute__((ext_vector_type(16))) __bf16 v16bf;
typedef __attribute__((ext_vector_type(8)))  float  v8f;

struct Frag32B { uint4 lo, hi; };

static __device__ __forceinline__ v16bf make_frag(const __bf16* p0, const __bf16* p1) {
  Frag32B t;
  t.lo = *(const uint4*)p0;
  t.hi = *(const uint4*)p1;
  return __builtin_bit_cast(v16bf, t);
}

// gfx1250 async global->LDS copy (16B), tracked by ASYNCcnt.
// vdst is a byte offset from the wave's LDS base; so plain byte offsets into dynamic LDS work.
static __device__ __forceinline__ void async_copy_b128(unsigned lds_off, const void* gptr) {
  asm volatile("global_load_async_to_lds_b128 %0, %1, off"
               :: "v"(lds_off), "v"(gptr)
               : "memory");
}
static __device__ __forceinline__ void wait_async() {
  asm volatile("s_wait_asynccnt 0x0" ::: "memory");
}

// ---------------------------------------------------------------- conversion
__global__ void f32_to_bf16_kernel(const float* __restrict__ s,
                                   __bf16* __restrict__ d, int n) {
  int i = blockIdx.x * blockDim.x + threadIdx.x;
  if (i < n) d[i] = (__bf16)s[i];
}

// W [K x N] f32 -> WT [N x K] bf16  (one-time; lets GEMM B-tiles be b128 copies)
__global__ void f32_to_bf16_T_kernel(const float* __restrict__ W,
                                     __bf16* __restrict__ WT, int K, int N) {
  int i = blockIdx.x * blockDim.x + threadIdx.x;   // index into WT
  if (i >= N * K) return;
  int n = i / K, k = i % K;
  WT[i] = (__bf16)W[(size_t)k * N + n];
}

// ---------------------------------------------------------------- WMMA GEMM
// C[M x N] (f32) = A[M x K] (bf16 rowmajor) @ WT[N x K] (bf16, pre-transposed) + bias
// 256 threads = 8 waves; block tile 128(M) x 128(N); wave tile 32x64
// (2 A-frags x 4 B-frags -> 8 WMMA, B reused 2x, A reused 4x).
// K stepped by 32; double-buffered async global->LDS staging overlapped with WMMA.
// Dynamic LDS: A 2x8KB @0, W 2x8KB @16384 => 32768 bytes.
__global__ __launch_bounds__(256)
void gemm_bf16_wmma(const __bf16* __restrict__ A, const __bf16* __restrict__ WT,
                    const float* __restrict__ bias, float* __restrict__ C,
                    int M, int N, int K) {
  extern __shared__ char smem[];
  const int tid   = threadIdx.x;
  const int lane  = tid & 31;
  const int wid   = tid >> 5;
  const int wm    = wid >> 1;            // 0..3 : 32-row strip
  const int wn    = wid & 1;             // 0..1 : 64-col strip
  const int m0    = blockIdx.y * 128;
  const int n0    = blockIdx.x * 128;
  const int lrow  = lane & 15;
  const int half8 = (lane >> 4) * 8;

  auto issue_tile = [&](int k0, int buf) {
    // A tile 128x32: 512 vectors, 2 per thread
    // W tile 128x32 (as [n][k]): 512 vectors, 2 per thread
#pragma unroll
    for (int s = 0; s < 2; ++s) {
      int idx = tid + s * 256;
      int r = idx >> 2;                  // 0..127
      int c = (idx & 3) * 8;             // {0,8,16,24}
      async_copy_b128((unsigned)(buf * 8192 + (r * 32 + c) * 2),
                      A + (size_t)(m0 + r) * K + k0 + c);
      async_copy_b128((unsigned)(16384 + buf * 8192 + (r * 32 + c) * 2),
                      WT + (size_t)(n0 + r) * K + k0 + c);
    }
  };
  auto aptr = [&](int buf) { return (__bf16*)smem + buf * 4096; };
  auto wptr = [&](int buf) { return (__bf16*)smem + 8192 + buf * 4096; };

  v8f acc[2][4];
  v8f zero = {0.f, 0.f, 0.f, 0.f, 0.f, 0.f, 0.f, 0.f};
#pragma unroll
  for (int i = 0; i < 2; ++i)
#pragma unroll
    for (int j = 0; j < 4; ++j) acc[i][j] = zero;

  issue_tile(0, 0);
  wait_async();
  __syncthreads();

  int buf = 0;
  for (int k0 = 0; k0 < K; k0 += 32) {
    if (k0 + 32 < K) issue_tile(k0 + 32, buf ^ 1);  // prefetch next tile (async)

    const __bf16* ab = aptr(buf);
    const __bf16* wb = wptr(buf);
    v16bf afrag[2];
#pragma unroll
    for (int i = 0; i < 2; ++i) {
      int row = 32 * wm + 16 * i + lrow;
      afrag[i] = make_frag(ab + row * 32 + half8, ab + row * 32 + half8 + 16);
    }
#pragma unroll
    for (int nt = 0; nt < 4; ++nt) {
      int nrow = 64 * wn + 16 * nt + lrow;
      v16bf bfrag = make_frag(wb + nrow * 32 + half8, wb + nrow * 32 + half8 + 16);
#pragma unroll
      for (int i = 0; i < 2; ++i)
        acc[i][nt] = __builtin_amdgcn_wmma_f32_16x16x32_bf16(
            false, afrag[i], false, bfrag, (short)0, acc[i][nt], false, false);
    }

    wait_async();
    __syncthreads();
    buf ^= 1;
  }

#pragma unroll
  for (int i = 0; i < 2; ++i) {
    const int mbase = m0 + 32 * wm + 16 * i + half8;
#pragma unroll
    for (int nt = 0; nt < 4; ++nt) {
      int ncol = n0 + 64 * wn + 16 * nt + lrow;
      float bv = bias[ncol];
#pragma unroll
      for (int r = 0; r < 8; ++r)
        C[(size_t)(mbase + r) * N + ncol] = acc[i][nt][r] + bv;
    }
  }
}

// ---------------------------------------------------------------- combine
// Qt,Qs pre-scaled by 1/sqrt(D); K1 = kt + lam_ts*ks; K2 = lam_st*kt + lam_ss*ks.
// Qt/Qs/K1/K2 head-major [B,H,L,D]; Vt TRANSPOSED to [B,H,D,L] so the flash
// kernel's V tiles are straight b128 async copies.
__global__ void combine_kernel(const float* __restrict__ qkvt,
                               const float* __restrict__ qss,
                               const float* __restrict__ lam_ts,
                               const float* __restrict__ lam_st,
                               const float* __restrict__ lam_ss,
                               __bf16* __restrict__ Qt, __bf16* __restrict__ Qs,
                               __bf16* __restrict__ K1, __bf16* __restrict__ K2,
                               __bf16* __restrict__ VtT) {
  int i = blockIdx.x * blockDim.x + threadIdx.x;
  if (i >= R_ * E_) return;
  int row = i / E_;           // b*L + l
  int e   = i % E_;
  int b = row >> 11, l = row & (L_ - 1);
  int h = e >> 6,    d = e & 63;
  size_t hidx = (((size_t)(b * H_ + h) * L_) + l) * D_ + d;
  size_t vidx = (((size_t)(b * H_ + h) * D_) + d) * L_ + l;

  float qt = qkvt[(size_t)row * 3 * E_ + e];
  float kt = qkvt[(size_t)row * 3 * E_ + E_ + e];
  float vt = qkvt[(size_t)row * 3 * E_ + 2 * E_ + e];
  float qs = qss[(size_t)row * 2 * E_ + e];
  float ks = qss[(size_t)row * 2 * E_ + E_ + e];

  const float scale = 0.125f;  // 1/sqrt(64)
  Qt[hidx] = (__bf16)(qt * scale);
  Qs[hidx] = (__bf16)(qs * scale);
  K1[hidx] = (__bf16)(kt + lam_ts[0] * ks);
  K2[hidx] = (__bf16)(lam_st[0] * kt + lam_ss[0] * ks);
  VtT[vidx] = (__bf16)vt;
}

// ---------------------------------------------------------------- flash attn
// grid (B*H, L/64); block 128 = 4 waves; each wave owns 16 query rows.
// S = Qt@K1^T + Qs@K2^T (scale pre-folded), online softmax, O += P@V.
// Double-buffered async staging of K1/K2/V^T tiles.
// Dynamic LDS: K1 2x8KB @0, K2 2x8KB @16384, V^T 2x8KB @32768, P 8KB @49152
//   => 57344 bytes.
__global__ __launch_bounds__(128)
void flash_attn_kernel(const __bf16* __restrict__ Qt, const __bf16* __restrict__ Qs,
                       const __bf16* __restrict__ K1, const __bf16* __restrict__ K2,
                       const __bf16* __restrict__ VtT, __bf16* __restrict__ Y) {
  extern __shared__ char smem[];
  const int tid   = threadIdx.x;
  const int lane  = tid & 31;
  const int wid   = tid >> 5;          // 0..3
  const int bh    = blockIdx.x;        // b*H + h
  const int q0    = blockIdx.y * 64;
  const int lrow  = lane & 15;
  const int half8 = (lane >> 4) * 8;

  const size_t base  = (size_t)bh * L_ * D_;   // for [B,H,L,D] tensors
  const size_t vbase = (size_t)bh * D_ * L_;   // for V^T [B,H,D,L]

  auto k1p = [&](int buf) { return (__bf16*)smem + buf * 4096; };
  auto k2p = [&](int buf) { return (__bf16*)smem + 8192 + buf * 4096; };
  auto vtp = [&](int buf) { return (__bf16*)smem + 16384 + buf * 4096; };
  __bf16* pbuf = (__bf16*)smem + 24576 + wid * 1024;   // this wave's 16x64 P tile

  auto issue_tile = [&](int kb, int buf) {
#pragma unroll
    for (int s = 0; s < 4; ++s) {
      int idx = tid + s * 128;           // 0..511
      int r = idx >> 3;                  // row 0..63 (key for K1/K2, d for V^T)
      int c = (idx & 7) * 8;             // col 0..56
      unsigned loff = (unsigned)((r * 64 + c) * 2);
      async_copy_b128(buf * 8192 + loff,         K1 + base + (size_t)(kb + r) * D_ + c);
      async_copy_b128(16384 + buf * 8192 + loff, K2 + base + (size_t)(kb + r) * D_ + c);
      async_copy_b128(32768 + buf * 8192 + loff, VtT + vbase + (size_t)r * L_ + kb + c);
    }
  };

  // preload Q fragments (D=64 -> two K-chunks of 32) for Qt and Qs
  v16bf qtf[2], qsf[2];
  {
    const __bf16* rt = Qt + base + (size_t)(q0 + 16 * wid + lrow) * D_;
    const __bf16* rs = Qs + base + (size_t)(q0 + 16 * wid + lrow) * D_;
#pragma unroll
    for (int kc = 0; kc < 2; ++kc) {
      qtf[kc] = make_frag(rt + 32 * kc + half8, rt + 32 * kc + half8 + 16);
      qsf[kc] = make_frag(rs + 32 * kc + half8, rs + 32 * kc + half8 + 16);
    }
  }

  v8f zero = {0.f, 0.f, 0.f, 0.f, 0.f, 0.f, 0.f, 0.f};
  v8f oacc[4];
#pragma unroll
  for (int i = 0; i < 4; ++i) oacc[i] = zero;
  float mrow[8], lsum[8];
#pragma unroll
  for (int r = 0; r < 8; ++r) { mrow[r] = -1e30f; lsum[r] = 0.f; }

  issue_tile(0, 0);
  wait_async();
  __syncthreads();

  int buf = 0;
  for (int kb = 0; kb < L_; kb += 64) {
    if (kb + 64 < L_) issue_tile(kb + 64, buf ^ 1);  // prefetch next key block

    // ---- scores S (16x64) = Qt@K1^T + Qs@K2^T
    const __bf16* k1b = k1p(buf);
    const __bf16* k2b = k2p(buf);
    v8f sacc[4];
#pragma unroll
    for (int i = 0; i < 4; ++i) sacc[i] = zero;
#pragma unroll
    for (int kc = 0; kc < 2; ++kc) {
#pragma unroll
      for (int nt = 0; nt < 4; ++nt) {
        int key = 16 * nt + lrow;
        v16bf b1 = make_frag(k1b + key * 64 + 32 * kc + half8,
                             k1b + key * 64 + 32 * kc + half8 + 16);
        sacc[nt] = __builtin_amdgcn_wmma_f32_16x16x32_bf16(
            false, qtf[kc], false, b1, (short)0, sacc[nt], false, false);
        v16bf b2 = make_frag(k2b + key * 64 + 32 * kc + half8,
                             k2b + key * 64 + 32 * kc + half8 + 16);
        sacc[nt] = __builtin_amdgcn_wmma_f32_16x16x32_bf16(
            false, qsf[kc], false, b2, (short)0, sacc[nt], false, false);
      }
    }

    // ---- online softmax (rows 0-7 in lanes 0-15, rows 8-15 in lanes 16-31)
    float rmax[8];
#pragma unroll
    for (int r = 0; r < 8; ++r)
      rmax[r] = fmaxf(fmaxf(sacc[0][r], sacc[1][r]),
                      fmaxf(sacc[2][r], sacc[3][r]));
#pragma unroll
    for (int m = 8; m >= 1; m >>= 1)
#pragma unroll
      for (int r = 0; r < 8; ++r)
        rmax[r] = fmaxf(rmax[r], __shfl_xor(rmax[r], m, 16));

    float alpha[8];
#pragma unroll
    for (int r = 0; r < 8; ++r) {
      float nm = fmaxf(mrow[r], rmax[r]);
      alpha[r] = __expf(mrow[r] - nm);
      mrow[r]  = nm;
    }

    float rsum[8];
#pragma unroll
    for (int r = 0; r < 8; ++r) rsum[r] = 0.f;
#pragma unroll
    for (int nt = 0; nt < 4; ++nt) {
#pragma unroll
      for (int r = 0; r < 8; ++r) {
        float p = __expf(sacc[nt][r] - mrow[r]);
        sacc[nt][r] = p;
        rsum[r] += p;
        oacc[nt][r] *= alpha[r];
      }
    }
#pragma unroll
    for (int m = 8; m >= 1; m >>= 1)
#pragma unroll
      for (int r = 0; r < 8; ++r)
        rsum[r] += __shfl_xor(rsum[r], m, 16);
#pragma unroll
    for (int r = 0; r < 8; ++r) lsum[r] = lsum[r] * alpha[r] + rsum[r];

    // ---- stage P as bf16 in this wave's LDS slab for A-fragment reload
#pragma unroll
    for (int nt = 0; nt < 4; ++nt)
#pragma unroll
      for (int r = 0; r < 8; ++r)
        pbuf[(r + half8) * 64 + 16 * nt + lrow] = (__bf16)sacc[nt][r];
    __syncthreads();

    // ---- O += P @ V (keys are the K dim, chunks of 32); V^T LDS is [d][key]
    const __bf16* vb = vtp(buf);
#pragma unroll
    for (int kc = 0; kc < 2; ++kc) {
      v16bf pf = make_frag(pbuf + lrow * 64 + 32 * kc + half8,
                           pbuf + lrow * 64 + 32 * kc + half8 + 16);
#pragma unroll
      for (int nt = 0; nt < 4; ++nt) {
        int d = 16 * nt + lrow;
        v16bf bv = make_frag(vb + d * 64 + 32 * kc + half8,
                             vb + d * 64 + 32 * kc + half8 + 16);
        oacc[nt] = __builtin_amdgcn_wmma_f32_16x16x32_bf16(
            false, pf, false, bv, (short)0, oacc[nt], false, false);
      }
    }

    wait_async();
    __syncthreads();
    buf ^= 1;
  }

  // ---- normalize, write Y [R, E] bf16 (cols h*64 + d)
  const int b    = bh / H_;
  const int h    = bh % H_;
  const int rowb = b * L_ + q0 + 16 * wid + half8;
#pragma unroll
  for (int nt = 0; nt < 4; ++nt) {
    int dcol = h * 64 + 16 * nt + lrow;
#pragma unroll
    for (int r = 0; r < 8; ++r) {
      float o = oacc[nt][r] / lsum[r];
      Y[(size_t)(rowb + r) * E_ + dcol] = (__bf16)o;
    }
  }
}

// ---------------------------------------------------------------- launch
extern "C" void kernel_launch(void* const* d_in, const int* in_sizes, int n_in,
                              void* d_out, int out_size, void* d_ws, size_t ws_size,
                              hipStream_t stream) {
  (void)in_sizes; (void)n_in; (void)out_size; (void)ws_size;
  const float* xt     = (const float*)d_in[0];
  const float* xs     = (const float*)d_in[1];
  const float* Wt     = (const float*)d_in[2];
  const float* bt     = (const float*)d_in[3];
  const float* Ws     = (const float*)d_in[4];
  const float* bs     = (const float*)d_in[5];
  const float* Wc     = (const float*)d_in[6];
  const float* bc     = (const float*)d_in[7];
  const float* lam_ts = (const float*)d_in[8];
  const float* lam_st = (const float*)d_in[9];
  const float* lam_ss = (const float*)d_in[10];

  // workspace bump allocator (256B aligned)
  char* ws = (char*)d_ws;
  size_t off = 0;
  auto alloc = [&](size_t bytes) {
    void* p = ws + off;
    off += (bytes + 255) & ~(size_t)255;
    return p;
  };
  __bf16* Xtb  = (__bf16*)alloc((size_t)R_ * E_ * 2);
  __bf16* Xsb  = (__bf16*)alloc((size_t)R_ * E_ * 2);
  __bf16* WtT  = (__bf16*)alloc((size_t)E_ * 3 * E_ * 2);   // [3E][E]
  __bf16* WsT  = (__bf16*)alloc((size_t)E_ * 2 * E_ * 2);   // [2E][E]
  __bf16* WcT  = (__bf16*)alloc((size_t)E_ * E_ * 2);       // [E][E]
  float*  qkvt = (float*)alloc((size_t)R_ * 3 * E_ * 4);
  float*  qss  = (float*)alloc((size_t)R_ * 2 * E_ * 4);
  __bf16* Qtb  = (__bf16*)alloc((size_t)R_ * E_ * 2);
  __bf16* Qsb  = (__bf16*)alloc((size_t)R_ * E_ * 2);
  __bf16* K1b  = (__bf16*)alloc((size_t)R_ * E_ * 2);
  __bf16* K2b  = (__bf16*)alloc((size_t)R_ * E_ * 2);
  __bf16* VtT  = (__bf16*)alloc((size_t)R_ * E_ * 2);       // [B,H,D,L]
  __bf16* Ybf  = (__bf16*)alloc((size_t)R_ * E_ * 2);

  f32_to_bf16_kernel<<<(R_ * E_ + 255) / 256, 256, 0, stream>>>(xt, Xtb, R_ * E_);
  f32_to_bf16_kernel<<<(R_ * E_ + 255) / 256, 256, 0, stream>>>(xs, Xsb, R_ * E_);
  f32_to_bf16_T_kernel<<<(E_ * 3 * E_ + 255) / 256, 256, 0, stream>>>(Wt, WtT, E_, 3 * E_);
  f32_to_bf16_T_kernel<<<(E_ * 2 * E_ + 255) / 256, 256, 0, stream>>>(Ws, WsT, E_, 2 * E_);
  f32_to_bf16_T_kernel<<<(E_ * E_ + 255) / 256, 256, 0, stream>>>(Wc, WcT, E_, E_);

  const size_t gemm_lds = 32768;
  // QKV_t = xt @ Wt + bt       (4096 x 3072)
  gemm_bf16_wmma<<<dim3(3 * E_ / 128, R_ / 128), 256, gemm_lds, stream>>>(
      Xtb, WtT, bt, qkvt, R_, 3 * E_, E_);
  // [qs|ks] = xs @ Ws + bs     (4096 x 2048)
  gemm_bf16_wmma<<<dim3(2 * E_ / 128, R_ / 128), 256, gemm_lds, stream>>>(
      Xsb, WsT, bs, qss, R_, 2 * E_, E_);

  combine_kernel<<<(R_ * E_ + 255) / 256, 256, 0, stream>>>(
      qkvt, qss, lam_ts, lam_st, lam_ss, Qtb, Qsb, K1b, K2b, VtT);

  flash_attn_kernel<<<dim3(B_ * H_, L_ / 64), 128, 57344, stream>>>(
      Qtb, Qsb, K1b, K2b, VtT, Ybf);

  // out = Y @ Wc + bc          (4096 x 1024) f32
  gemm_bf16_wmma<<<dim3(E_ / 128, R_ / 128), 256, gemm_lds, stream>>>(
      Ybf, WcT, bc, (float*)d_out, R_, E_, E_);
}
